// GEM_36034775613526
// MI455X (gfx1250) — matrix operand order repaired and checked by reading, not compiled, not run on hardware
//
#include <hip/hip_runtime.h>
#include <math.h>

// ---------------- types ----------------
typedef __bf16 bf16;
typedef __attribute__((ext_vector_type(16))) bf16  v16bf;
typedef __attribute__((ext_vector_type(8)))  bf16  v8bf;
typedef __attribute__((ext_vector_type(8)))  float v8f;
typedef unsigned short ushort_t;
typedef unsigned int   uint_t;

#define DIN   256
#define DOUT  128
#define CCLS  2
#define HOP   2
#define DEVN  3

// ---------------- helpers ----------------
__device__ __forceinline__ ushort_t f32_to_bf16_rne(float f) {
    uint_t u = __float_as_uint(f);
    uint_t r = u + 0x7FFFu + ((u >> 16) & 1u);   // round-to-nearest-even
    return (ushort_t)(r >> 16);
}

// ---------------- elementwise kernels ----------------
__global__ void k_zero_f32(float* p, long n) {
    long i = (long)blockIdx.x * blockDim.x + threadIdx.x;
    if (i < n) p[i] = 0.0f;
}

__global__ void k_copy_f32(const float* src, float* dst, long n) {
    long i = (long)blockIdx.x * blockDim.x + threadIdx.x;
    if (i < n) dst[i] = src[i];
}

__global__ void k_f32_to_bf16(const float* src, ushort_t* dst, long n) {
    long i = (long)blockIdx.x * blockDim.x + threadIdx.x;
    if (i < n) dst[i] = f32_to_bf16_rne(src[i]);
}

__global__ void k_sigmoid(const float* src, float* dst, long n) {
    long i = (long)blockIdx.x * blockDim.x + threadIdx.x;
    if (i < n) dst[i] = 1.0f / (1.0f + expf(-src[i]));
}

// ---------------- sparse aggregation (scatter-add) ----------------
// agg[dst[e], :] += vals[e] * h[src[e], :]   (DOUT=128 lanes per edge)
__global__ void k_scatter(const float* __restrict__ h,
                          const int*   __restrict__ src,
                          const int*   __restrict__ dst,
                          const float* __restrict__ vals,
                          float* __restrict__ agg, int E) {
    int e   = blockIdx.x * 2 + (threadIdx.x >> 7);   // 2 edges per 256-thread block
    int col = threadIdx.x & 127;
    if (e >= E) return;
    int   s = src[e];
    int   t = dst[e];
    float v = vals[e];
    atomicAdd(&agg[(size_t)t * DOUT + col], v * h[(size_t)s * DOUT + col]);
}

// ---------------- WMMA GEMM: C[m,n] (+)= alpha * A[m,k] * B[k,n] ----------------
// A: [mtiles*16, K] bf16 row-major ; B: [K, DOUT] bf16 row-major ; C: [mtiles*16, DOUT] f32
// One wave computes one 16x16 tile of C. K % 32 == 0. EXEC stays all-ones.
__global__ void k_gemm_wmma_bf16(const ushort_t* __restrict__ A_,
                                 const ushort_t* __restrict__ B_,
                                 float* __restrict__ C,
                                 const float* __restrict__ alpha_p,
                                 int mtiles, int K, int accumulate) {
    const int wave  = threadIdx.x >> 5;
    const int lane  = threadIdx.x & 31;
    const int mtile = blockIdx.x * (blockDim.x >> 5) + wave;
    if (mtile >= mtiles) return;                     // wave-uniform retire

    const int m0   = mtile * 16;
    const int n0   = blockIdx.y * 16;
    const int half = lane >> 4;
    const int l16  = lane & 15;

    const bf16* Ab = (const bf16*)A_;
    const bf16* Bb = (const bf16*)B_;

    v8f c = {};
    const int m = m0 + l16;

    for (int k0 = 0; k0 < K; k0 += 32) {
        // A fragment: lane holds row m; K = [k0+8*half .. +7] and [k0+16+8*half .. +7]
        const v8bf lo = *(const v8bf*)(Ab + (size_t)m * K + k0 + 8 * half);
        const v8bf hi = *(const v8bf*)(Ab + (size_t)m * K + k0 + 16 + 8 * half);
        v16bf a;
#pragma unroll
        for (int i = 0; i < 8; ++i) { a[i] = lo[i]; a[8 + i] = hi[i]; }

        // B fragment: lane holds column n0+l16; K = [k0+8*half .. +7] and [k0+16+8*half .. +7]
        v16bf b;
#pragma unroll
        for (int i = 0; i < 8; ++i) {
            b[i]     = Bb[(size_t)(k0 + 8 * half + i) * DOUT + n0 + l16];
            b[8 + i] = Bb[(size_t)(k0 + 16 + 8 * half + i) * DOUT + n0 + l16];
        }

        // (neg_a, A, neg_b, B, c_mod, C, reuse_a, reuse_b)
        c = __builtin_amdgcn_wmma_f32_16x16x32_bf16(false, a, false, b,
                                                    (short)0, c, false, false);
    }

    const float alpha = alpha_p ? *alpha_p : 1.0f;
    const int n = n0 + l16;
#pragma unroll
    for (int r = 0; r < 8; ++r) {
        const size_t idx = (size_t)(m0 + r + 8 * half) * DOUT + n;
        if (accumulate) C[idx] += alpha * c[r];
        else            C[idx] = c[r];
    }
}

// ---------------- masked loss + accuracy ----------------
__global__ void k_mask_loss_acc(const float* __restrict__ h,
                                const float* __restrict__ label,
                                const int*   __restrict__ idx_mask,
                                const float* __restrict__ Wc,
                                const float* __restrict__ bvec,
                                const float* __restrict__ u,
                                float* __restrict__ accum, int M) {
    int i = blockIdx.x * blockDim.x + threadIdx.x;
    if (i >= M) return;
    int node = idx_mask[i];
    const float* row = h + (size_t)node * DOUT;

    float score = 0.0f;
    float l0 = bvec[0], l1 = bvec[1];
#pragma unroll 4
    for (int k = 0; k < DOUT; ++k) {
        float v = row[k];
        score += v * u[k];
        l0 += v * Wc[k * CCLS + 0];
        l1 += v * Wc[k * CCLS + 1];
    }

    float loss = 0.0f;
#pragma unroll
    for (int c = 0; c < CCLS; ++c) {
        float z  = label[(size_t)node * CCLS + c] * score;
        float ls = fminf(z, 0.0f) - log1pf(expf(-fabsf(z)));  // stable log_sigmoid
        loss -= ls;
    }
    int pred = (l1 > l0) ? 1 : 0;
    int lab  = (label[(size_t)node * CCLS + 1] > label[(size_t)node * CCLS + 0]) ? 1 : 0;

    atomicAdd(&accum[0], loss);
    atomicAdd(&accum[1], (pred == lab) ? 1.0f : 0.0f);
}

__global__ void k_finalize(const float* accum, float* out, float invM) {
    out[0] = accum[0];
    out[1] = accum[1] * invM;
}

// ---------------- launcher ----------------
extern "C" void kernel_launch(void* const* d_in, const int* in_sizes, int n_in,
                              void* d_out, int out_size, void* d_ws, size_t ws_size,
                              hipStream_t stream) {
    const float* x        = (const float*)d_in[0];
    const float* label    = (const float*)d_in[1];
    const int*   idx_mask = (const int*)  d_in[2];
    const int*   src      = (const int*)  d_in[3];
    const int*   dst      = (const int*)  d_in[4];
    const float* vals     = (const float*)d_in[5];
    const float* h0       = (const float*)d_in[6];
    const float* Ws       = (const float*)d_in[7];
    const float* Vs       = (const float*)d_in[8];
    const float* alphas   = (const float*)d_in[9];
    const float* Wc       = (const float*)d_in[10];
    const float* bvec     = (const float*)d_in[11];
    const float* u        = (const float*)d_in[12];
    float* out = (float*)d_out;

    const int N = in_sizes[0] / DIN;          // 50000
    const int M = in_sizes[2];                // 10000
    const int E = in_sizes[3] / DEVN;         // 500000
    const long NH = (long)N * DOUT;           // h elements
    const long NX = (long)N * DIN;

    // workspace carve (256B aligned)
    char* p = (char*)d_ws;
    auto carve = [&](size_t bytes) { char* r = p; p += (bytes + 255) & ~(size_t)255; return (void*)r; };
    ushort_t* xb    = (ushort_t*)carve((size_t)NX * 2);
    float*    h     = (float*)   carve((size_t)NH * 4);
    float*    h1    = (float*)   carve((size_t)NH * 4);
    float*    agg   = (float*)   carve((size_t)NH * 4);
    ushort_t* aggb  = (ushort_t*)carve((size_t)NH * 2);
    ushort_t* Wb    = (ushort_t*)carve((size_t)DIN * DOUT * 2);
    ushort_t* Vb    = (ushort_t*)carve((size_t)DOUT * DOUT * 2);
    float*    accum = (float*)   carve(2 * 4);

    const int TB = 256;
    auto blocks = [&](long n) { return (unsigned)((n + TB - 1) / TB); };

    const int mtiles = N / 16;                           // 3125
    const dim3 gemm_grid((mtiles + 3) / 4, DOUT / 16);   // 4 waves/block, 8 N-tiles
    const dim3 gemm_block(128);

    // one-time per call
    k_f32_to_bf16<<<blocks(NX), TB, 0, stream>>>(x, xb, NX);
    k_copy_f32   <<<blocks(NH), TB, 0, stream>>>(h0, h, NH);

    for (int hop = 0; hop < HOP; ++hop) {
        k_f32_to_bf16<<<blocks((long)DIN * DOUT), TB, 0, stream>>>(
            Ws + (size_t)hop * DIN * DOUT, Wb, (long)DIN * DOUT);
        k_f32_to_bf16<<<blocks((long)DOUT * DOUT), TB, 0, stream>>>(
            Vs + (size_t)hop * DOUT * DOUT, Vb, (long)DOUT * DOUT);

        // h1 = x @ W
        k_gemm_wmma_bf16<<<gemm_grid, gemm_block, 0, stream>>>(
            xb, Wb, h1, nullptr, mtiles, DIN, /*accumulate=*/0);

        for (int d = 0; d < DEVN; ++d) {
            k_zero_f32<<<blocks(NH), TB, 0, stream>>>(agg, NH);
            k_scatter<<<(E + 1) / 2, 256, 0, stream>>>(
                h, src + (size_t)d * E, dst + (size_t)d * E, vals + (size_t)d * E, agg, E);
            k_f32_to_bf16<<<blocks(NH), TB, 0, stream>>>(agg, aggb, NH);
            // h1 += alpha[hop][d] * (agg @ V)
            k_gemm_wmma_bf16<<<gemm_grid, gemm_block, 0, stream>>>(
                aggb, Vb, h1, alphas + (size_t)hop * DEVN + d, mtiles, DOUT, /*accumulate=*/1);
        }

        // h = sigmoid(h1)
        k_sigmoid<<<blocks(NH), TB, 0, stream>>>(h1, h, NH);
    }

    k_zero_f32<<<1, 32, 0, stream>>>(accum, 2);
    k_mask_loss_acc<<<blocks(M), TB, 0, stream>>>(h, label, idx_mask, Wc, bvec, u, accum, M);
    k_finalize<<<1, 1, 0, stream>>>(accum, out, 1.0f / (float)M);
}
